// GraphTransformerLayer_35785667510650
// MI455X (gfx1250) — matrix operand order repaired and checked by reading, not compiled, not run on hardware
//
#include <hip/hip_runtime.h>

// ---------------------------------------------------------------------------
// Graph Transformer layer for MI455X (gfx1250, wave32, WMMA).
// All GEMM-shaped math uses V_WMMA_F32_16X16X32_F16 (f16 in, f32 accumulate).
// Attention is flash-style: never materializes the 8x4096x4096 score tensor.
// GEMM waves compute 16x64 tiles (4 independent accumulators) to amortize
// A-fragment loads and avoid WMMA->WMMA RAW hazard NOPs.
// ---------------------------------------------------------------------------

typedef _Float16 half_t;
typedef __attribute__((ext_vector_type(16))) _Float16 v16h;
typedef __attribute__((ext_vector_type(8)))  _Float16 v8h;
typedef __attribute__((ext_vector_type(8)))  float    v8f;

#define CAT16(lo, hi) __builtin_shufflevector((lo), (hi), 0,1,2,3,4,5,6,7,8,9,10,11,12,13,14,15)

constexpr int kN    = 4096;
constexpr int kHID  = 512;
constexpr int kH    = 8;
constexpr int kHD   = 64;
constexpr int kPE   = 64;
constexpr int kQKIN = kHID + kPE;   // 576

// ---------------------------------------------------------------------------
// Elementwise packing / precision conversion
// ---------------------------------------------------------------------------
__global__ __launch_bounds__(256) void pack_inputs(
    const float* __restrict__ x, const float* __restrict__ peQ,
    const float* __restrict__ peK,
    half_t* __restrict__ xq16, half_t* __restrict__ xk16, half_t* __restrict__ x16)
{
    int tid = blockIdx.x * blockDim.x + threadIdx.x;
    if (tid >= kN * kQKIN) return;
    int row = tid / kQKIN, col = tid - row * kQKIN;
    if (col < kHID) {
        float vx = x[(size_t)row * kHID + col];
        xq16[tid] = (half_t)vx;
        xk16[tid] = (half_t)vx;
        x16[(size_t)row * kHID + col] = (half_t)vx;
    } else {
        xq16[tid] = (half_t)peQ[(size_t)row * kPE + (col - kHID)];
        xk16[tid] = (half_t)peK[(size_t)row * kPE + (col - kHID)];
    }
}

// W[z][K][NC] (f32) -> WT[z][NC][K] (f16): B-fragment rows become contiguous.
__global__ __launch_bounds__(256) void convert_transpose(
    const float* __restrict__ W, half_t* __restrict__ WT, int K, int NC)
{
    const size_t z = blockIdx.z;
    const float* Wz = W + z * (size_t)K * NC;
    half_t* WTz = WT + z * (size_t)NC * K;
    int tid = blockIdx.x * blockDim.x + threadIdx.x;
    if (tid >= K * NC) return;
    int n = tid / K, k = tid - n * K;
    WTz[tid] = (half_t)Wz[(size_t)k * NC + n];
}

// ---------------------------------------------------------------------------
// Per-wave 16x64-tile WMMA GEMM: C = A[M,K] * BT[NC,K]^T (+ epilogue).
// 4 independent 16x16 accumulators: A fragment reused 4x per k-step and the
// 4 WMMAs per step are independent (no RAW hazard NOPs).
// MODE 0: out f16 [z][M,NC]   = C + bias                      (Q/K proj)
// MODE 1: out f16 [z][NC,M]   = C + bias (transposed store)   (V -> V^T)
// MODE 2: out f32 x1 = aux0(x) + relu(C+bias)*rsqrt(aux1=deg) (Wc + residual)
// MODE 3: out f16 = C + bias                                  (W1)
// MODE 4: out f32 = aux0(x1n) + C + bias                      (W2 + residual)
// ---------------------------------------------------------------------------
template <int MODE>
__global__ __launch_bounds__(256) void gemm_wmma(
    const half_t* __restrict__ A, const half_t* __restrict__ BT,
    const float* __restrict__ bias, void* __restrict__ outp,
    const float* __restrict__ aux0, const float* __restrict__ aux1,
    int M, int NC, int K, long long btZ, long long outZ, long long biasZ)
{
    const int z = blockIdx.z;
    BT   += (size_t)z * btZ;
    bias += (size_t)z * biasZ;

    const int lane = threadIdx.x & 31;
    const int wave = threadIdx.x >> 5;
    const int hi   = lane >> 4;          // 0: lanes 0-15, 1: lanes 16-31
    const int nl   = lane & 15;

    const int tilesN   = NC >> 6;                 // 64-wide column tiles
    const int numTiles = (M >> 4) * tilesN;
    const int w = blockIdx.x * 8 + wave;
    if (w >= numTiles) return;           // uniform per wave: EXEC stays all-ones
    const int tm = w / tilesN, tn = w - tm * tilesN;

    const half_t* arp = A + (size_t)(tm * 16 + nl) * K;        // A row (m = nl)
    const half_t* brp[4];
#pragma unroll
    for (int g = 0; g < 4; ++g)                                // B col (n = nl)
        brp[g] = BT + (size_t)(tn * 64 + g * 16 + nl) * K;

    v8f c[4] = {v8f{}, v8f{}, v8f{}, v8f{}};
    for (int k0 = 0; k0 < K; k0 += 32) {
        // A fragment: 16 contiguous halves at K = k0+hi*16
        v16h a = *(const v16h*)(arp + k0 + hi * 16);
#pragma unroll
        for (int g = 0; g < 4; ++g) {
            // B fragment: 16 contiguous halves at K = k0+hi*16
            v16h b = *(const v16h*)(brp[g] + k0 + hi * 16);
            c[g] = __builtin_amdgcn_wmma_f32_16x16x32_f16(
                       false, a, false, b, (short)0, c[g], false, false);
        }
    }

#pragma unroll
    for (int g = 0; g < 4; ++g) {
#pragma unroll
        for (int r = 0; r < 8; ++r) {
            const int row = tm * 16 + r + 8 * hi;   // C-frag row mapping
            const int col = tn * 64 + g * 16 + nl;
            float v = c[g][r] + bias[col];
            if constexpr (MODE == 0) {
                half_t* o = (half_t*)outp + (size_t)z * outZ;
                o[(size_t)row * NC + col] = (half_t)v;
            } else if constexpr (MODE == 1) {
                half_t* o = (half_t*)outp + (size_t)z * outZ;
                o[(size_t)col * M + row] = (half_t)v;     // V^T layout [HD, N]
            } else if constexpr (MODE == 2) {
                float* o = (float*)outp;
                v = fmaxf(v, 0.0f);
                o[(size_t)row * NC + col] =
                    aux0[(size_t)row * NC + col] + v * rsqrtf(aux1[row]);
            } else if constexpr (MODE == 3) {
                half_t* o = (half_t*)outp;
                o[(size_t)row * NC + col] = (half_t)v;
            } else {
                float* o = (float*)outp;
                o[(size_t)row * NC + col] = aux0[(size_t)row * NC + col] + v;
            }
        }
    }
}

// ---------------------------------------------------------------------------
// Flash attention: block = 8 waves, each wave owns 16 queries of one head.
// Per 32-key step: 4 WMMA for S = Q K^T (two independent accumulators,
// interleaved to dodge WMMA->WMMA hazards), LDS transpose of P, 4 WMMA for PV.
// ---------------------------------------------------------------------------
__global__ __launch_bounds__(256) void flash_attn(
    const half_t* __restrict__ Q16, const half_t* __restrict__ K16,
    const half_t* __restrict__ VT16, half_t* __restrict__ heads16)
{
    __shared__ __align__(32) half_t ldsP[8][16][32];   // per-wave P tile (8 KB)

    const int lane = threadIdx.x & 31;
    const int wave = threadIdx.x >> 5;
    const int hi   = lane >> 4;
    const int nl   = lane & 15;
    const int h    = blockIdx.y;
    const int q0   = (blockIdx.x * 8 + wave) * 16;

    const half_t* Qh  = Q16  + (size_t)h * kN * kHD;
    const half_t* Kh  = K16  + (size_t)h * kN * kHD;
    const half_t* VTh = VT16 + (size_t)h * kHD * kN;

    // Q as two A-fragments (16 queries x 64 dims), loaded once
    v16h aq[2];
#pragma unroll
    for (int f = 0; f < 2; ++f) {
        const half_t* p = Qh + (size_t)(q0 + nl) * kHD + f * 32 + hi * 16;
        aq[f] = *(const v16h*)p;
    }

    v8f  o[4] = {v8f{}, v8f{}, v8f{}, v8f{}};   // 16 queries x 64 dims
    float mrow[8], lrow[8];
#pragma unroll
    for (int r = 0; r < 8; ++r) { mrow[r] = -3.0e38f; lrow[r] = 0.0f; }

    for (int kb = 0; kb < kN / 32; ++kb) {
        // ---- S = Q K^T : two 16x16 key tiles, interleaved accumulation ----
        const half_t* kp0 = Kh + (size_t)(kb * 32 + nl) * kHD + hi * 16;
        const half_t* kp1 = kp0 + (size_t)16 * kHD;
        v16h bk00 = *(const v16h*)(kp0);        // tile 0, dims  0..31
        v16h bk01 = *(const v16h*)(kp0 + 32);   // tile 0, dims 32..63
        v16h bk10 = *(const v16h*)(kp1);        // tile 1, dims  0..31
        v16h bk11 = *(const v16h*)(kp1 + 32);
        v8f s0 = {}, s1 = {};
        s0 = __builtin_amdgcn_wmma_f32_16x16x32_f16(false, aq[0], false, bk00, (short)0, s0, false, false);
        s1 = __builtin_amdgcn_wmma_f32_16x16x32_f16(false, aq[0], false, bk10, (short)0, s1, false, false);
        s0 = __builtin_amdgcn_wmma_f32_16x16x32_f16(false, aq[1], false, bk01, (short)0, s0, false, false);
        s1 = __builtin_amdgcn_wmma_f32_16x16x32_f16(false, aq[1], false, bk11, (short)0, s1, false, false);

        // ---- online softmax (rows live in 16-lane halves) ----
#pragma unroll
        for (int r = 0; r < 8; ++r) {
            float v0 = s0[r] * 0.125f;            // 1/sqrt(64)
            float v1 = s1[r] * 0.125f;
            float mt = fmaxf(v0, v1);
            mt = fmaxf(mt, __shfl_xor(mt, 1, 32));
            mt = fmaxf(mt, __shfl_xor(mt, 2, 32));
            mt = fmaxf(mt, __shfl_xor(mt, 4, 32));
            mt = fmaxf(mt, __shfl_xor(mt, 8, 32));
            float mnew = fmaxf(mrow[r], mt);
            float corr = __expf(mrow[r] - mnew);
            float p0 = __expf(v0 - mnew);
            float p1 = __expf(v1 - mnew);
            float ps = p0 + p1;
            ps += __shfl_xor(ps, 1, 32);
            ps += __shfl_xor(ps, 2, 32);
            ps += __shfl_xor(ps, 4, 32);
            ps += __shfl_xor(ps, 8, 32);
            lrow[r] = lrow[r] * corr + ps;
            mrow[r] = mnew;
#pragma unroll
            for (int g = 0; g < 4; ++g) o[g][r] *= corr;
            const int m = r + 8 * hi;
            ldsP[wave][m][nl]      = (half_t)p0;
            ldsP[wave][m][16 + nl] = (half_t)p1;
        }
        __syncthreads();   // uniform trip count; C-layout -> A-layout via LDS

        // ---- reload P as A-fragment (16 queries x 32 keys) ----
        v16h ap = *(const v16h*)&ldsP[wave][nl][hi * 16];

        // ---- O += P V  (V^T rows contiguous over keys; 4 independent) ----
#pragma unroll
        for (int g = 0; g < 4; ++g) {
            v16h bv = *(const v16h*)(VTh + (size_t)(g * 16 + nl) * kN +
                                     kb * 32 + hi * 16);
            o[g] = __builtin_amdgcn_wmma_f32_16x16x32_f16(
                       false, ap, false, bv, (short)0, o[g], false, false);
        }
    }

    // normalized heads, laid out [N, H*HD] for the Wc GEMM
#pragma unroll
    for (int r = 0; r < 8; ++r) {
        const int m = r + 8 * hi;
        const float inv = 1.0f / lrow[r];
        size_t base = (size_t)(q0 + m) * kHID + (size_t)h * kHD;
#pragma unroll
        for (int g = 0; g < 4; ++g)
            heads16[base + g * 16 + nl] = (half_t)(o[g][r] * inv);
    }
}

// ---------------------------------------------------------------------------
// BatchNorm (training mode, biased variance): column stats over N rows
// ---------------------------------------------------------------------------
__global__ __launch_bounds__(256) void bn_stats(
    const float* __restrict__ X, float* __restrict__ mu, float* __restrict__ rs,
    int rows, int cols)
{
    __shared__ float ss[256], sq[256];
    const int c = blockIdx.x;
    float s = 0.0f, q = 0.0f;
    for (int r = threadIdx.x; r < rows; r += blockDim.x) {
        float v = X[(size_t)r * cols + c];
        s += v; q += v * v;
    }
    ss[threadIdx.x] = s; sq[threadIdx.x] = q;
    __syncthreads();
    for (int st = 128; st > 0; st >>= 1) {
        if (threadIdx.x < st) {
            ss[threadIdx.x] += ss[threadIdx.x + st];
            sq[threadIdx.x] += sq[threadIdx.x + st];
        }
        __syncthreads();
    }
    if (threadIdx.x == 0) {
        float m = ss[0] / rows;
        float var = sq[0] / rows - m * m;
        mu[c] = m;
        rs[c] = rsqrtf(var + 1e-5f);
    }
}

__global__ __launch_bounds__(256) void bn_apply(
    const float* __restrict__ X, const float* __restrict__ mu,
    const float* __restrict__ rs, const float* __restrict__ g,
    const float* __restrict__ b, float* __restrict__ Yf,
    half_t* __restrict__ Yh, int total, int cols)
{
    int i = blockIdx.x * blockDim.x + threadIdx.x;
    if (i >= total) return;
    int c = i % cols;
    float v = (X[i] - mu[c]) * rs[c] * g[c] + b[c];
    if (Yf) Yf[i] = v;
    if (Yh) Yh[i] = (half_t)v;
}

// ---------------------------------------------------------------------------
extern "C" void kernel_launch(void* const* d_in, const int* in_sizes, int n_in,
                              void* d_out, int out_size, void* d_ws, size_t ws_size,
                              hipStream_t stream)
{
    const float* x   = (const float*)d_in[0];
    const float* peQ = (const float*)d_in[1];
    const float* peK = (const float*)d_in[2];
    const float* deg = (const float*)d_in[3];
    const float* WQ  = (const float*)d_in[4];
    const float* bQ  = (const float*)d_in[5];
    const float* WK  = (const float*)d_in[6];
    const float* bK  = (const float*)d_in[7];
    const float* WV  = (const float*)d_in[8];
    const float* bV  = (const float*)d_in[9];
    const float* Wc  = (const float*)d_in[10];
    const float* bc  = (const float*)d_in[11];
    const float* g1  = (const float*)d_in[12];
    const float* be1 = (const float*)d_in[13];
    const float* W1  = (const float*)d_in[14];
    const float* b1  = (const float*)d_in[15];
    const float* W2  = (const float*)d_in[16];
    const float* b2  = (const float*)d_in[17];
    const float* g2  = (const float*)d_in[18];
    const float* be2 = (const float*)d_in[19];
    float* out = (float*)d_out;

    char* base = (char*)d_ws;
    size_t off = 0;
    auto alloc = [&](size_t bytes) -> void* {
        off = (off + 255) & ~(size_t)255;
        void* p = base + off;
        off += bytes;
        return p;
    };

    half_t* xq16   = (half_t*)alloc((size_t)kN * kQKIN * 2);
    half_t* xk16   = (half_t*)alloc((size_t)kN * kQKIN * 2);
    half_t* x16    = (half_t*)alloc((size_t)kN * kHID * 2);
    half_t* WQT    = (half_t*)alloc((size_t)kH * kHD * kQKIN * 2);
    half_t* WKT    = (half_t*)alloc((size_t)kH * kHD * kQKIN * 2);
    half_t* WVT    = (half_t*)alloc((size_t)kH * kHD * kHID * 2);
    half_t* WcT    = (half_t*)alloc((size_t)kHID * kHID * 2);
    half_t* W1T    = (half_t*)alloc((size_t)kHID * kHID * 2);
    half_t* W2T    = (half_t*)alloc((size_t)kHID * kHID * 2);
    half_t* Q16    = (half_t*)alloc((size_t)kH * kN * kHD * 2);
    half_t* K16    = (half_t*)alloc((size_t)kH * kN * kHD * 2);
    half_t* VT16   = (half_t*)alloc((size_t)kH * kHD * kN * 2);
    half_t* heads16= (half_t*)alloc((size_t)kN * kHID * 2);
    float*  x1     = (float*) alloc((size_t)kN * kHID * 4);
    float*  x1n    = (float*) alloc((size_t)kN * kHID * 4);
    half_t* x1n16  = (half_t*)alloc((size_t)kN * kHID * 2);
    half_t* h116   = (half_t*)alloc((size_t)kN * kHID * 2);
    float*  y      = (float*) alloc((size_t)kN * kHID * 4);
    float*  mu1    = (float*) alloc(kHID * 4);
    float*  rs1    = (float*) alloc(kHID * 4);
    float*  mu2    = (float*) alloc(kHID * 4);
    float*  rs2    = (float*) alloc(kHID * 4);

    const dim3 blk(256);
    const int total = kN * kHID;

    // 1. pack inputs to f16
    pack_inputs<<<dim3((kN * kQKIN + 255) / 256), blk, 0, stream>>>(
        x, peQ, peK, xq16, xk16, x16);

    // 2. weight transpose + f16 convert
    convert_transpose<<<dim3((kQKIN * kHD + 255) / 256, 1, kH), blk, 0, stream>>>(WQ, WQT, kQKIN, kHD);
    convert_transpose<<<dim3((kQKIN * kHD + 255) / 256, 1, kH), blk, 0, stream>>>(WK, WKT, kQKIN, kHD);
    convert_transpose<<<dim3((kHID * kHD + 255) / 256, 1, kH), blk, 0, stream>>>(WV, WVT, kHID, kHD);
    convert_transpose<<<dim3((kHID * kHID + 255) / 256, 1, 1), blk, 0, stream>>>(Wc, WcT, kHID, kHID);
    convert_transpose<<<dim3((kHID * kHID + 255) / 256, 1, 1), blk, 0, stream>>>(W1, W1T, kHID, kHID);
    convert_transpose<<<dim3((kHID * kHID + 255) / 256, 1, 1), blk, 0, stream>>>(W2, W2T, kHID, kHID);

    // 3. Q/K/V projections (per-head WMMA GEMMs); 16x64 tiles -> 256 waves/head
    const int wavesQK = (kN / 16) * (kHD / 64);   // 256
    gemm_wmma<0><<<dim3(wavesQK / 8, 1, kH), blk, 0, stream>>>(
        xq16, WQT, bQ, Q16, nullptr, nullptr,
        kN, kHD, kQKIN, (long long)kHD * kQKIN, (long long)kN * kHD, kHD);
    gemm_wmma<0><<<dim3(wavesQK / 8, 1, kH), blk, 0, stream>>>(
        xk16, WKT, bK, K16, nullptr, nullptr,
        kN, kHD, kQKIN, (long long)kHD * kQKIN, (long long)kN * kHD, kHD);
    gemm_wmma<1><<<dim3(wavesQK / 8, 1, kH), blk, 0, stream>>>(
        x16, WVT, bV, VT16, nullptr, nullptr,
        kN, kHD, kHID, (long long)kHD * kHID, (long long)kHD * kN, kHD);

    // 4. flash attention -> heads16 [N, HID]
    flash_attn<<<dim3(kN / 128, kH), blk, 0, stream>>>(Q16, K16, VT16, heads16);

    // 5. output proj + ReLU + deg-scaled residual -> x1
    const int wavesBig = (kN / 16) * (kHID / 64);   // 2048
    gemm_wmma<2><<<dim3(wavesBig / 8, 1, 1), blk, 0, stream>>>(
        heads16, WcT, bc, x1, x, deg, kN, kHID, kHID, 0, 0, 0);

    // 6. BatchNorm 1
    bn_stats<<<dim3(kHID), blk, 0, stream>>>(x1, mu1, rs1, kN, kHID);
    bn_apply<<<dim3((total + 255) / 256), blk, 0, stream>>>(
        x1, mu1, rs1, g1, be1, x1n, x1n16, total, kHID);

    // 7. FFN: h1 = x1n @ W1 + b1 ; y = x1n + h1 @ W2 + b2
    gemm_wmma<3><<<dim3(wavesBig / 8, 1, 1), blk, 0, stream>>>(
        x1n16, W1T, b1, h116, nullptr, nullptr, kN, kHID, kHID, 0, 0, 0);
    gemm_wmma<4><<<dim3(wavesBig / 8, 1, 1), blk, 0, stream>>>(
        h116, W2T, b2, y, x1n, nullptr, kN, kHID, kHID, 0, 0, 0);

    // 8. BatchNorm 2 -> output
    bn_stats<<<dim3(kHID), blk, 0, stream>>>(y, mu2, rs2, kN, kHID);
    bn_apply<<<dim3((total + 255) / 256), blk, 0, stream>>>(
        y, mu2, rs2, g2, be2, out, nullptr, total, kHID);

    (void)in_sizes; (void)n_in; (void)out_size; (void)ws_size;
}
